// SimpleLSTM_57226144252096
// MI455X (gfx1250) — matrix operand order repaired and checked by reading, not compiled, not run on hardware
//
#include <hip/hip_runtime.h>
#include <math.h>

// ---------------------------------------------------------------------------
// 3-layer LSTM, fused persistent pipeline for MI455X (gfx1250, wave32, WMMA)
//   block 0 = layer 1, block 1 = layer 2, block 2 = layer 3 + final linear
//   layer l -> layer l+1 via L2-resident ring buffer in d_ws (64-step depth)
//   per step per layer: gates[128x512] = [z_t | h_{t-1}] (128x256) x B (256x512)
//   computed with V_WMMA_F32_16X16X32_F16, c-state held in VGPRs all 2048 steps
//
// Round-2 changes: weight tiles stored per-(jt) contiguously so the whole GEMM
// runs off 3 base addresses + immediate ds offsets (kills scratch spills seen
// in round 1); h-store swizzle hoisted to linear form; prefetch on x/w stream.
// ---------------------------------------------------------------------------

typedef _Float16 v16h __attribute__((ext_vector_type(16)));
typedef float    v8f  __attribute__((ext_vector_type(8)));

#define HDIM   128
#define LSEQ   2048
#define NBATCH 128
#define DIN    64
#define RING   64                       // ring depth (power of 2)
#define FRAG   512                      // f16 elems per 16x16x32 operand frag
#define W_F16  (256 * FRAG)             // 32 n-tiles * 8 k-frags  = 256 KB
#define A_F16  (32 * FRAG)              // 8 m-tiles * 4 k-frags   =  32 KB
#define SMEM_BYTES ((W_F16 + 2 * A_F16) * 2)   // 320 KB LDS per WGP

// A-operand (16-bit) WMMA fragment-order address for element (m, k).
// Per ISA 7.12.2: lanes 0-15 hold K {0..7,16..23}, lanes 16-31 hold {8..15,24..31}.
__device__ __forceinline__ int swizA(int m, int k) {
  int kk  = k & 31;
  int hi  = (kk >> 3) & 1;
  int idx = (kk & 7) + ((kk >= 16) ? 8 : 0);
  int lane = (m & 15) + (hi << 4);
  return ((((m >> 4) << 2) + (k >> 5)) << 9) + (lane << 4) + idx;
}

__device__ __forceinline__ float sigm_f(float x) { return 1.0f / (1.0f + __expf(-x)); }
__device__ __forceinline__ float tanh_f(float x) {
  float e = __expf(2.0f * x);
  return (e - 1.0f) / (e + 1.0f);
}

struct Params {
  const float* x;
  const float* w;
  const float* Wih[3];
  const float* Whh[3];
  const float* bih[3];
  const float* bhh[3];
  const float* Wlin;
  const float* blin;
  float*       out;
  _Float16*    ring0;          // RING slots x 16384 f16, A-swizzled h of layer 1
  _Float16*    ring1;          // same for layer 2
  unsigned int* flags;         // [0]=prod0 [16]=prod1 [32]=cons0 [48]=cons1
};

__global__ __launch_bounds__(1024, 1)
void lstm3_pipeline(Params p) {
  extern __shared__ _Float16 smem[];
  _Float16* Wl   = smem;                    // B operand, per-jt contiguous order
  _Float16* zbuf = smem + W_F16;            // A, K 0..127  (layer input)
  _Float16* hbuf = smem + W_F16 + A_F16;    // A, K 128..255 (h_{t-1})

  const int layer = blockIdx.x;
  const int tid   = threadIdx.x;
  const int lane  = tid & 31;
  const int wave  = tid >> 5;
  const int jt    = wave & 7;               // hidden-unit tile 0..7
  const int mtp   = (wave >> 3) << 1;       // batch tiles mtp, mtp+1

  const float* Wih = p.Wih[layer];
  const float* Whh = p.Whh[layer];
  _Float16* ringIn  = (layer == 1) ? p.ring0 : p.ring1;
  _Float16* ringOut = (layer == 0) ? p.ring0 : p.ring1;
  unsigned int* prodUp = p.flags + (layer > 0 ? (layer - 1) * 16 : 0);
  unsigned int* prodMe = p.flags + layer * 16;
  unsigned int* consMe = p.flags + 32 + layer * 16;
  unsigned int* consUp = p.flags + 32 + (layer > 0 ? (layer - 1) * 16 : 0);

  // ---- stage combined weight matrix B[k][n] = (k<128 ? Wih[n][k] : Whh[n][k-128])
  // into LDS in B-fragment order, with tile groups permuted so wave jt's four
  // gate tiles (n-tiles g*8+jt, g=0..3) are contiguous:
  //   fragIdx = ((jt*4 + g) * 8) + kt
  for (int i = 0; i < 128; ++i) {
    int pos    = (tid << 7) + i;
    int frag   = pos >> 9;
    int within = pos & 511;
    int l      = within >> 4;
    int idx    = within & 15;
    int grp    = frag >> 3;                 // jt*4 + g
    int kt     = frag & 7;
    int jt_l   = grp >> 2;
    int g_l    = grp & 3;
    int n      = ((g_l << 3) + jt_l) * 16 + (l & 15);
    int k      = (kt << 5) + ((idx >> 3) << 4) + ((l >> 4) << 3) + (idx & 7);
    float v = (k < HDIM) ? Wih[n * HDIM + k] : Whh[n * HDIM + (k - HDIM)];
    Wl[pos] = (_Float16)v;
  }
  // h_{-1} = 0
  for (int i = tid; i < A_F16; i += 1024) hbuf[i] = (_Float16)0.0f;

  // per-lane gate biases (lane <-> column j is fixed by D layout)
  float bias[4];
#pragma unroll
  for (int g = 0; g < 4; ++g) {
    int n = g * HDIM + (jt << 4) + (lane & 15);
    bias[g] = p.bih[layer][n] + p.bhh[layer][n];
  }

  // ---- the three (and only three) LDS base addresses the GEMM needs
  const _Float16* wB = Wl   + (jt << 14) + (lane << 4);          // jt*32 frags
  const _Float16* zB = zbuf + ((mtp << 2) << 9) + (lane << 4);
  const _Float16* hB = hbuf + ((mtp << 2) << 9) + (lane << 4);

  // ---- hoisted h-store swizzle: pos(mi, e) = hbase[mi] + 16*e
  const int j    = (jt << 4) + (lane & 15);
  const int jj   = j & 31;
  const int hij  = (jj >> 3) & 1;
  const int idxj = (jj & 7) + ((jj >= 16) ? 8 : 0);
  const int hl   = lane >> 4;
  int hbase[2];
#pragma unroll
  for (int mi = 0; mi < 2; ++mi)
    hbase[mi] = ((((mtp + mi) << 2) + (j >> 5)) << 9) +
                (((hl << 3) + (hij << 4)) << 4) + idxj;

  float creg[2][8] = {};                    // cell state: lives in VGPRs all run

  // layer-0 prefetch registers: z_t = concat(x_t, w_t), one step ahead
  float rx[16];
  const int zm  = tid >> 3;
  const int zk0 = (tid & 7) << 4;
  const bool zx = (zk0 < DIN);
  const float* zsrc = zx ? (p.x + zm * LSEQ * DIN + zk0)
                         : (p.w + zm * LSEQ * DIN + (zk0 - DIN));
  if (layer == 0) {
#pragma unroll
    for (int i = 0; i < 16; ++i) rx[i] = zsrc[i];
  }
  __syncthreads();

  for (int t = 0; t < LSEQ; ++t) {
    // ---- flow control: consumer waits for producer; producer honors ring depth
    if (tid == 0) {
      if (layer > 0) {
        while (__hip_atomic_load(prodUp, __ATOMIC_ACQUIRE,
                                 __HIP_MEMORY_SCOPE_AGENT) < (unsigned)(t + 1))
          __builtin_amdgcn_s_sleep(1);
      }
      if (layer < 2 && t >= RING) {
        while (__hip_atomic_load(consMe, __ATOMIC_ACQUIRE,
                                 __HIP_MEMORY_SCOPE_AGENT) < (unsigned)(t - RING + 1))
          __builtin_amdgcn_s_sleep(1);
      }
    }
    __syncthreads();

    const int slot = t & (RING - 1);
    _Float16* ringSlotOut = ringOut + (size_t)slot * A_F16;
    if (layer == 0) {
      // commit prefetched z_t (f32->f16, swizzled scatter), then issue t+1 loads
#pragma unroll
      for (int i = 0; i < 16; ++i) zbuf[swizA(zm, zk0 + i)] = (_Float16)rx[i];
      if (t + 1 < LSEQ) {
        const float* s1 = zsrc + (t + 1) * DIN;
#pragma unroll
        for (int i = 0; i < 16; ++i) rx[i] = s1[i];
      }
      if (t + 2 < LSEQ)
        __builtin_prefetch(zsrc + (t + 2) * DIN, 0, 1);   // global_prefetch_b8
    } else {
      // ring slot is already in A-fragment order: straight 32 B/thread copy
      const uint4* src = (const uint4*)(ringIn + (size_t)slot * A_F16);
      uint4* dst = (uint4*)zbuf;
      uint4 v0 = src[(tid << 1)];
      uint4 v1 = src[(tid << 1) + 1];
      dst[(tid << 1)]     = v0;
      dst[(tid << 1) + 1] = v1;
    }
    __syncthreads();

    // ---- GEMM: each wave owns 2 batch tiles x 4 gate tiles at column tile jt.
    // All operands via immediate offsets from wB/zB/hB (no address registers).
    v8f acc[2][4] = {};
#pragma unroll
    for (int kt = 0; kt < 8; ++kt) {
      const _Float16* ab = (kt < 4) ? zB : hB;
      const int akt = kt & 3;
      v16h a0 = *(const v16h*)(ab + (akt << 9));
      v16h a1 = *(const v16h*)(ab + ((4 + akt) << 9));
#pragma unroll
      for (int g = 0; g < 4; ++g) {
        v16h b = *(const v16h*)(wB + (((g << 3) + kt) << 9));
        acc[0][g] = __builtin_amdgcn_wmma_f32_16x16x32_f16(
            false, a0, false, b, (short)0, acc[0][g], false, false);
        acc[1][g] = __builtin_amdgcn_wmma_f32_16x16x32_f16(
            false, a1, false, b, (short)0, acc[1][g], false, false);
      }
    }
    __syncthreads();   // all reads of h_{t-1}/z_t done -> safe to overwrite hbuf

    // ---- LSTM cell update (D layout: VGPR e -> row, lane -> column)
#pragma unroll
    for (int mi = 0; mi < 2; ++mi) {
      const int pb = hbase[mi];
#pragma unroll
      for (int e = 0; e < 8; ++e) {
        float gi = sigm_f(acc[mi][0][e] + bias[0]);
        float gf = sigm_f(acc[mi][1][e] + bias[1]);
        float gg = tanh_f(acc[mi][2][e] + bias[2]);
        float go = sigm_f(acc[mi][3][e] + bias[3]);
        float c  = gf * creg[mi][e] + gi * gg;
        creg[mi][e] = c;
        float h = go * tanh_f(c);
        const int pos = pb + (e << 4);
        _Float16 hh = (_Float16)h;
        hbuf[pos] = hh;
        if (layer < 2) ringSlotOut[pos] = hh;
      }
    }
    __syncthreads();

    if (tid == 0) {
      if (layer < 2) {
        __threadfence();
        __hip_atomic_store(prodMe, (unsigned)(t + 1), __ATOMIC_RELEASE,
                           __HIP_MEMORY_SCOPE_AGENT);
      }
      if (layer > 0)
        __hip_atomic_store(consUp, (unsigned)(t + 1), __ATOMIC_RELEASE,
                           __HIP_MEMORY_SCOPE_AGENT);
    }
  }

  // ---- final linear on h_{L-1} of layer 3: out[m] = b + sum_j h[m][j]*Wlin[j]
  if (layer == 2) {
    __syncthreads();
    if (tid < NBATCH) {
      float s = p.blin[0];
#pragma unroll 4
      for (int jj2 = 0; jj2 < HDIM; ++jj2)
        s += (float)hbuf[swizA(tid, jj2)] * p.Wlin[jj2];
      p.out[tid] = s;
    }
  }
}

extern "C" void kernel_launch(void* const* d_in, const int* in_sizes, int n_in,
                              void* d_out, int out_size, void* d_ws, size_t ws_size,
                              hipStream_t stream) {
  (void)in_sizes; (void)n_in; (void)out_size; (void)ws_size;
  Params p;
  p.x      = (const float*)d_in[0];
  p.w      = (const float*)d_in[1];
  p.Wih[0] = (const float*)d_in[2];
  p.Whh[0] = (const float*)d_in[3];
  p.bih[0] = (const float*)d_in[4];
  p.bhh[0] = (const float*)d_in[5];
  p.Wih[1] = (const float*)d_in[6];
  p.Whh[1] = (const float*)d_in[7];
  p.bih[1] = (const float*)d_in[8];
  p.bhh[1] = (const float*)d_in[9];
  p.Wih[2] = (const float*)d_in[10];
  p.Whh[2] = (const float*)d_in[11];
  p.bih[2] = (const float*)d_in[12];
  p.bhh[2] = (const float*)d_in[13];
  p.Wlin   = (const float*)d_in[14];
  p.blin   = (const float*)d_in[15];
  p.out    = (float*)d_out;

  unsigned char* ws = (unsigned char*)d_ws;
  p.flags = (unsigned int*)ws;
  p.ring0 = (_Float16*)(ws + 4096);
  p.ring1 = (_Float16*)(ws + 4096 + (size_t)RING * A_F16 * 2);

  // re-arm the pipeline flags on every call (graph-capture safe memset node)
  hipMemsetAsync(ws, 0, 4096, stream);

  hipFuncSetAttribute(reinterpret_cast<const void*>(lstm3_pipeline),
                      hipFuncAttributeMaxDynamicSharedMemorySize, SMEM_BYTES);
  lstm3_pipeline<<<dim3(3), dim3(1024), SMEM_BYTES, stream>>>(p);
}
